// NewAdditiveAttention_71519795413668
// MI455X (gfx1250) — compile-verified, matrix operand order checked by reading, not run on hardware
//
#include <hip/hip_runtime.h>
#include <math.h>

#define BATCH 32
#define SEQ   2048
#define HLEN  1024
#define VLEN  1024
#define HID   1024

typedef unsigned short ushort_t;
typedef __attribute__((ext_vector_type(16))) __bf16 v16bf;
typedef __attribute__((ext_vector_type(8)))  float  v8f;

union Frag16 {
    v16bf v;
    uint4 q[2];
};

__device__ __forceinline__ ushort_t f32_to_bf16(float f) {
    unsigned int u = __float_as_uint(f);
    unsigned int r = u + 0x7FFFu + ((u >> 16) & 1u);   // round-to-nearest-even
    return (ushort_t)(r >> 16);
}

// ---------------------------------------------------------------------------
// Kernel 1: repack U_w (HID x VLEN, f32) into bf16 B-fragment layout:
// Ufrag[((kc*HID) + d)*32 + kl] = bf16(U_w[d, kc*32 + kl])
// so a B fragment is two contiguous 16B loads per lane.
// ---------------------------------------------------------------------------
__global__ __launch_bounds__(256) void uprep_kernel(const float* __restrict__ Uw,
                                                    ushort_t* __restrict__ Ufrag) {
    int idx = blockIdx.x * 256 + threadIdx.x;       // 0 .. HID*VLEN-1
    if (idx >= HID * VLEN) return;
    int kl = idx & 31;
    int d  = (idx >> 5) & (HID - 1);
    int kc = idx >> 15;                              // /(32*HID)
    float f = Uw[(size_t)d * VLEN + kc * 32 + kl];
    Ufrag[idx] = f32_to_bf16(f);
}

// ---------------------------------------------------------------------------
// Kernel 2: h_proj = h @ W_w^T + W_b   (32 x 1024), fp32 — tiny
// ---------------------------------------------------------------------------
__global__ __launch_bounds__(256) void hproj_kernel(const float* __restrict__ h,
                                                    const float* __restrict__ Ww,
                                                    const float* __restrict__ Wb,
                                                    float* __restrict__ hproj) {
    int idx = blockIdx.x * 256 + threadIdx.x;       // b*HID + d
    if (idx >= BATCH * HID) return;
    int b = idx >> 10;
    int d = idx & (HID - 1);
    const float* hr = h + (size_t)b * HLEN;
    const float* wr = Ww + (size_t)d * HLEN;
    float s = Wb[d];
    for (int k = 0; k < HLEN; k += 4) {
        float4 hv = *(const float4*)(hr + k);
        float4 wv = *(const float4*)(wr + k);
        s += hv.x * wv.x + hv.y * wv.y + hv.z * wv.z + hv.w * wv.w;
    }
    hproj[idx] = s;
}

// ---------------------------------------------------------------------------
// Kernel 3 (main): fused V_proj GEMM (bf16 WMMA) + tanh + w-dot -> e (32 x 2048)
// Grid: (BATCH, SEQ/32); block: 256 threads = 8 waves.
// Each block: 32 seq rows (2 row-tiles of 16); each wave owns 8 d-tiles of 16.
// B fragments are reused across both row-tiles (2 WMMAs per B load).
// ---------------------------------------------------------------------------
__global__ __launch_bounds__(256) void wmma_e_kernel(const float* __restrict__ V,
                                                     const ushort_t* __restrict__ Ufrag,
                                                     const float* __restrict__ hproj,
                                                     const float* __restrict__ ww,
                                                     float* __restrict__ e_out) {
    __shared__ __align__(16) ushort_t Asm[32 * 32];   // 32 rows x 32 K, bf16 (2KB)
    __shared__ float hp_sm[HID];
    __shared__ float w_sm[HID];
    __shared__ float e_sm[32];

    const int b    = blockIdx.x;
    const int s0   = blockIdx.y * 32;
    const int tid  = threadIdx.x;
    const int wave = tid >> 5;
    const int lane = tid & 31;
    const int col  = lane & 15;          // M (for A) / N (for B/C)
    const int bk   = (lane >> 4) * 8;    // K-half base: 0 or 8

    for (int i = tid; i < HID; i += 256) {
        hp_sm[i] = hproj[(size_t)b * HID + i];
        w_sm[i]  = ww[i];
    }
    if (tid < 32) e_sm[tid] = 0.0f;

    v8f acc0[8], acc1[8];                // [d-tile] for row-tiles 0 and 1
    #pragma unroll
    for (int j = 0; j < 8; ++j)
        #pragma unroll
        for (int g = 0; g < 8; ++g) { acc0[j][g] = 0.0f; acc1[j][g] = 0.0f; }

    const float* Vbase = V + ((size_t)b * SEQ + s0) * VLEN;

    for (int kc = 0; kc < VLEN / 32; ++kc) {
        __syncthreads();   // previous-iteration A reads complete
        // Stage A tile: 32 rows x 32 K = 1024 f32 -> bf16 LDS, 4 elems/thread
        {
            int e0 = tid * 4;                    // e0 == r*32 + k
            int r  = e0 >> 5;
            int k  = e0 & 31;
            float4 fv = *(const float4*)(Vbase + (size_t)r * VLEN + kc * 32 + k);
            uint2 p;
            p.x = (unsigned)f32_to_bf16(fv.x) | ((unsigned)f32_to_bf16(fv.y) << 16);
            p.y = (unsigned)f32_to_bf16(fv.z) | ((unsigned)f32_to_bf16(fv.w) << 16);
            *(uint2*)(&Asm[e0]) = p;
        }
        __syncthreads();

        // A fragments for both row tiles: K runs [bk,bk+8) and [bk+16,bk+24)
        Frag16 a0, a1;
        a0.q[0] = *(const uint4*)(&Asm[col * 32 + bk]);
        a0.q[1] = *(const uint4*)(&Asm[col * 32 + bk + 16]);
        a1.q[0] = *(const uint4*)(&Asm[(16 + col) * 32 + bk]);
        a1.q[1] = *(const uint4*)(&Asm[(16 + col) * 32 + bk + 16]);

        const bool pf = (kc + 1) < (VLEN / 32);
        #pragma unroll
        for (int j = 0; j < 8; ++j) {
            const int d0 = (wave * 8 + j) * 16;
            const ushort_t* bp = Ufrag + ((size_t)kc * HID + d0 + col) * 32;
            Frag16 bb;
            bb.q[0] = *(const uint4*)(bp + bk);
            bb.q[1] = *(const uint4*)(bp + bk + 16);
            if (pf) __builtin_prefetch(bp + (size_t)HID * 32, 0, 1);  // next K-chunk
            acc0[j] = __builtin_amdgcn_wmma_f32_16x16x32_bf16(
                false, a0.v, false, bb.v, (short)0, acc0[j], false, false);
            acc1[j] = __builtin_amdgcn_wmma_f32_16x16x32_bf16(
                false, a1.v, false, bb.v, (short)0, acc1[j], false, false);
        }
    }
    __syncthreads();

    // Epilogue: e[row] = sum_d w[d] * tanh(hproj[d] + Vproj[row,d])
    // C layout: VGPR g -> M = g + 8*(lane>=16), N = lane&15
    float rs0[8], rs1[8];
    #pragma unroll
    for (int g = 0; g < 8; ++g) { rs0[g] = 0.0f; rs1[g] = 0.0f; }
    #pragma unroll
    for (int j = 0; j < 8; ++j) {
        const int d = (wave * 8 + j) * 16 + col;
        const float hp = hp_sm[d];
        const float w  = w_sm[d];
        #pragma unroll
        for (int g = 0; g < 8; ++g) {
            rs0[g] += w * tanhf(hp + acc0[j][g]);
            rs1[g] += w * tanhf(hp + acc1[j][g]);
        }
    }
    // Reduce over the 16 lanes of each half (offsets < 16 stay inside half)
    #pragma unroll
    for (int off = 1; off < 16; off <<= 1)
        #pragma unroll
        for (int g = 0; g < 8; ++g) {
            rs0[g] += __shfl_xor(rs0[g], off, 32);
            rs1[g] += __shfl_xor(rs1[g], off, 32);
        }

    if (col == 0) {                       // lane 0 -> rows 0..7, lane 16 -> rows 8..15
        const int mbase = (lane >> 4) * 8;
        #pragma unroll
        for (int g = 0; g < 8; ++g) {
            atomicAdd(&e_sm[mbase + g],      rs0[g]);
            atomicAdd(&e_sm[16 + mbase + g], rs1[g]);
        }
    }
    __syncthreads();
    if (tid < 32)
        e_out[(size_t)b * SEQ + s0 + tid] = e_sm[tid];
}

// ---------------------------------------------------------------------------
// Kernel 4: softmax over seq per batch -> beta (written straight to d_out)
// ---------------------------------------------------------------------------
__global__ __launch_bounds__(256) void softmax_kernel(const float* __restrict__ e,
                                                      float* __restrict__ beta) {
    __shared__ float red[256];
    const int b = blockIdx.x, tid = threadIdx.x;
    const float* er = e + (size_t)b * SEQ;

    float m = -INFINITY;
    for (int s = tid; s < SEQ; s += 256) m = fmaxf(m, er[s]);
    red[tid] = m; __syncthreads();
    for (int o = 128; o > 0; o >>= 1) {
        if (tid < o) red[tid] = fmaxf(red[tid], red[tid + o]);
        __syncthreads();
    }
    m = red[0]; __syncthreads();

    float sum = 0.0f;
    for (int s = tid; s < SEQ; s += 256) sum += __expf(er[s] - m);
    red[tid] = sum; __syncthreads();
    for (int o = 128; o > 0; o >>= 1) {
        if (tid < o) red[tid] += red[tid + o];
        __syncthreads();
    }
    const float inv = 1.0f / red[0];
    for (int s = tid; s < SEQ; s += 256)
        beta[(size_t)b * SEQ + s] = __expf(er[s] - m) * inv;
}

// ---------------------------------------------------------------------------
// Kernel 5: zero result region; Kernel 6: result = beta @ V (atomic s-split)
// ---------------------------------------------------------------------------
__global__ __launch_bounds__(256) void zero_kernel(float* __restrict__ p, int n) {
    int i = blockIdx.x * 256 + threadIdx.x;
    if (i < n) p[i] = 0.0f;
}

__global__ __launch_bounds__(256) void result_kernel(const float* __restrict__ V,
                                                     const float* __restrict__ beta,
                                                     float* __restrict__ result) {
    const int SCH = 256;                 // seq chunk per block
    const int b  = blockIdx.x;
    const int sc = blockIdx.y;
    const int v  = blockIdx.z * 256 + threadIdx.x;
    const float* Vb = V + ((size_t)b * SEQ + (size_t)sc * SCH) * VLEN + v;
    const float* bb = beta + (size_t)b * SEQ + (size_t)sc * SCH;
    float acc = 0.0f;
    for (int s = 0; s < SCH; ++s)
        acc += bb[s] * Vb[(size_t)s * VLEN];
    atomicAdd(&result[(size_t)b * VLEN + v], acc);
}

// ---------------------------------------------------------------------------
extern "C" void kernel_launch(void* const* d_in, const int* in_sizes, int n_in,
                              void* d_out, int out_size, void* d_ws, size_t ws_size,
                              hipStream_t stream) {
    const float* h  = (const float*)d_in[0];   // (32, 1024)
    const float* V  = (const float*)d_in[1];   // (32, 2048, 1024)
    const float* Ww = (const float*)d_in[2];   // (1024, 1024)
    const float* Wb = (const float*)d_in[3];   // (1024,)
    const float* Uw = (const float*)d_in[4];   // (1024, 1024)
    const float* ww = (const float*)d_in[5];   // (1, 1024)

    float* result = (float*)d_out;                       // 32*1024
    float* beta   = (float*)d_out + BATCH * VLEN;        // 32*2048

    // Workspace layout
    ushort_t* Ufrag = (ushort_t*)d_ws;                                   // 2 MB
    float*    hproj = (float*)((char*)d_ws + (size_t)HID * VLEN * 2);    // 128 KB
    float*    e     = hproj + BATCH * HID;                               // 256 KB

    // 1. Repack U_w to bf16 fragment layout
    uprep_kernel<<<(HID * VLEN + 255) / 256, 256, 0, stream>>>(Uw, Ufrag);
    // 2. h_proj
    hproj_kernel<<<(BATCH * HID + 255) / 256, 256, 0, stream>>>(h, Ww, Wb, hproj);
    // 3. Fused WMMA projection + tanh + w-dot -> e
    wmma_e_kernel<<<dim3(BATCH, SEQ / 32), 256, 0, stream>>>(V, Ufrag, hproj, ww, e);
    // 4. Softmax -> beta (in output)
    softmax_kernel<<<BATCH, 256, 0, stream>>>(e, beta);
    // 5+6. result = beta @ V
    zero_kernel<<<(BATCH * VLEN + 255) / 256, 256, 0, stream>>>(result, BATCH * VLEN);
    result_kernel<<<dim3(BATCH, SEQ / 256, VLEN / 256), 256, 0, stream>>>(V, beta, result);
}
